// SwinTransformerBlock_72928544686437
// MI455X (gfx1250) — compile-verified, matrix operand order checked by reading, not run on hardware
//
#include <hip/hip_runtime.h>
#include <hip/hip_bf16.h>
#include <math.h>

typedef _Float16 half_t;
typedef __attribute__((ext_vector_type(16))) _Float16 v16h;
typedef __attribute__((ext_vector_type(8)))  _Float16 v8h;
typedef __attribute__((ext_vector_type(8)))  float    v8f;

#define WMMA_F16(A_, B_, C_) __builtin_amdgcn_wmma_f32_16x16x32_f16(false,(A_),false,(B_),(short)0,(C_),false,false)

#define TOK   65536   // total tokens = B*H*W
#define CDIM  512
#define NHEAD 16
#define HD    32

// ---------------------------------------------------------------------------
// Fragment loader: row-major LDS region, lane L reads row (rowbase + L%16),
// halves [koff..koff+7] and [koff+16..koff+23], koff = (L/16)*8.
// Matches the CDNA5 16-bit A-matrix 16x32 layout (and Bt[n][k] for B).
// ---------------------------------------------------------------------------
__device__ __forceinline__ v16h load_frag(const half_t* base, int rowbase, int ld, int lane) {
  int r    = rowbase + (lane & 15);
  int koff = (lane >> 4) * 8;
  const half_t* p = base + r * ld + koff;
  v8h lo = *(const v8h*)(p);
  v8h hi = *(const v8h*)(p + 16);
  v16h f;
#pragma unroll
  for (int i = 0; i < 8; ++i) { f[i] = lo[i]; f[i + 8] = hi[i]; }
  return f;
}

// ---------------------------------------------------------------------------
// Weight conversion / packing. All GEMM B-operands are stored TRANSPOSED
// (N-major, BT[n][k]) so the GEMM can stage them with contiguous 16B copies:
//   wqkvT [1536][512] = [q_w | k_w | v_w]^T,  bqkv = [q_b | 0 | v_b]
//   wprojT[512][512], wm1T[2048][512], wm2T[512][2048]
// ---------------------------------------------------------------------------
__global__ void pack_weights_kernel(const float* __restrict__ qw, const float* __restrict__ qb,
                                    const float* __restrict__ kw, const float* __restrict__ vw,
                                    const float* __restrict__ vb, const float* __restrict__ pw,
                                    const float* __restrict__ m1, const float* __restrict__ m2,
                                    half_t* __restrict__ wqkvT, float* __restrict__ bqkv,
                                    half_t* __restrict__ wprojT, half_t* __restrict__ wm1T,
                                    half_t* __restrict__ wm2T) {
  const long NQKV = 512L * 1536, NPRJ = 512L * 512, NM1 = 512L * 2048, NM2 = 2048L * 512;
  const long TOTAL = NQKV + NPRJ + NM1 + NM2 + 1536;
  for (long idx = blockIdx.x * (long)blockDim.x + threadIdx.x; idx < TOTAL;
       idx += (long)gridDim.x * blockDim.x) {
    long e = idx;
    if (e < NQKV) {
      long k = e / 1536, n = e % 1536;  // source (K x N) row-major
      float s = (n < 512) ? qw[k * 512 + n] : (n < 1024) ? kw[k * 512 + (n - 512)]
                                                         : vw[k * 512 + (n - 1024)];
      wqkvT[n * 512 + k] = (half_t)s; continue;
    }
    e -= NQKV;
    if (e < NPRJ) {
      long k = e / 512, n = e % 512;
      wprojT[n * 512 + k] = (half_t)pw[e]; continue;
    }
    e -= NPRJ;
    if (e < NM1) {
      long k = e / 2048, n = e % 2048;
      wm1T[n * 512 + k] = (half_t)m1[e]; continue;
    }
    e -= NM1;
    if (e < NM2) {
      long k = e / 512, n = e % 512;
      wm2T[n * 2048 + k] = (half_t)m2[e]; continue;
    }
    e -= NM2;
    bqkv[e] = (e < 512) ? qb[e] : (e < 1024) ? 0.0f : vb[e - 1024];
  }
}

// ---------------------------------------------------------------------------
// Cyclic shift (-4,-4) + 8x8 window partition, fp32 -> fp16.
// Token t = ((b*64 + win)*64 + n); source pixel = shifted coords.
// ---------------------------------------------------------------------------
__global__ void shift_partition_kernel(const float* __restrict__ x, half_t* __restrict__ xw) {
  int t = blockIdx.x;
  int b = t >> 12, rem = t & 4095;
  int win = rem >> 6, n = rem & 63;
  int wh = win >> 3, ww = win & 7;
  int r = n >> 3, c = n & 7;
  int sh = (wh * 8 + r + 4) & 63;
  int sw = (ww * 8 + c + 4) & 63;
  const float* xp = x + ((size_t)((b << 6 | sh) << 6 | sw)) * CDIM;
  half_t* op = xw + (size_t)t * CDIM;
  for (int i = threadIdx.x; i < CDIM; i += blockDim.x) op[i] = (half_t)xp[i];
}

// ---------------------------------------------------------------------------
// Continuous relative position bias MLP: 225 (dy,dx) rows -> 16 heads,
// output already 16*sigmoid(.) applied.
// ---------------------------------------------------------------------------
__device__ __forceinline__ float relnorm(int d) {
  float x = (float)d * (8.0f / 7.0f);
  float v = log2f(fabsf(x) + 1.0f) * (1.0f / 3.0f);
  return copysignf(v, x);
}

__global__ void cpb_kernel(const float* __restrict__ w1, const float* __restrict__ b1,
                           const float* __restrict__ w2, const float* __restrict__ b2,
                           float* __restrict__ tbl) {
  __shared__ float hid[512];
  int rIdx = blockIdx.x;          // 0..224
  int dy = rIdx / 15 - 7, dx = rIdx % 15 - 7;
  float f0 = relnorm(dx), f1 = relnorm(dy);
  for (int k = threadIdx.x; k < 512; k += blockDim.x)
    hid[k] = fmaxf(0.0f, f0 * w1[k] + f1 * w1[512 + k] + b1[k]);
  __syncthreads();
  if (threadIdx.x < NHEAD) {
    int h = threadIdx.x;
    float acc = b2[h];
    for (int k = 0; k < 512; ++k) acc += hid[k] * w2[k * NHEAD + h];
    tbl[rIdx * NHEAD + h] = 16.0f * __builtin_amdgcn_rcpf(1.0f + __expf(-acc));
  }
}

// ---------------------------------------------------------------------------
// Generic WMMA GEMM: C[M,N] = A[M,K](f16) * BT[N,K](f16, pre-transposed) + bias[N].
// Block = 64M x 128N tile, 128 threads = 4 waves, each wave 32Mx64N
// (2 A-frags x 4 B-frags = 8 wmma accums). Register double-buffered k-loop:
// next tile's global loads are issued while the current tile computes.
// MODE 0: store f16.  MODE 1: gelu(tanh) then store f16.  MODE 2: store f32.
// ---------------------------------------------------------------------------
template <int MODE>
__launch_bounds__(128)
__global__ void gemm_kernel(const half_t* __restrict__ A, const half_t* __restrict__ BT,
                            const float* __restrict__ bias, float* __restrict__ outf,
                            half_t* __restrict__ outh, int M, int N, int K) {
  __shared__ alignas(16) half_t As[64 * 32];
  __shared__ alignas(16) half_t Bts[128 * 32];   // Bt[n][k]
  const int tid = threadIdx.x;
  const int lane = tid & 31, wv = tid >> 5;
  const int wm = wv >> 1, wn = wv & 1;
  const int m0 = blockIdx.y * 64, n0 = blockIdx.x * 128;

  const int rowA = tid >> 1, khA = (tid & 1) * 16;
  const half_t* gpA = A  + (size_t)(m0 + rowA) * K + khA;  // 32B chunk per thread
  const half_t* gpB = BT + (size_t)(n0 + tid) * K;         // 64B row-chunk per thread

  v8f acc[2][4] = {};

  // Preload first k-tile into registers.
  v8h ra0 = *(const v8h*)(gpA);
  v8h ra1 = *(const v8h*)(gpA + 8);
  v8h rb0 = *(const v8h*)(gpB);
  v8h rb1 = *(const v8h*)(gpB + 8);
  v8h rb2 = *(const v8h*)(gpB + 16);
  v8h rb3 = *(const v8h*)(gpB + 24);

  for (int kt = 0; kt < K; kt += 32) {
    __syncthreads();   // previous compute done; LDS free to overwrite
    *(v8h*)&As[rowA * 32 + khA]     = ra0;
    *(v8h*)&As[rowA * 32 + khA + 8] = ra1;
    *(v8h*)&Bts[tid * 32]      = rb0;
    *(v8h*)&Bts[tid * 32 + 8]  = rb1;
    *(v8h*)&Bts[tid * 32 + 16] = rb2;
    *(v8h*)&Bts[tid * 32 + 24] = rb3;
    __syncthreads();

    if (kt + 32 < K) {  // issue next tile's loads; they overlap the WMMAs below
      ra0 = *(const v8h*)(gpA + kt + 32);
      ra1 = *(const v8h*)(gpA + kt + 40);
      rb0 = *(const v8h*)(gpB + kt + 32);
      rb1 = *(const v8h*)(gpB + kt + 40);
      rb2 = *(const v8h*)(gpB + kt + 48);
      rb3 = *(const v8h*)(gpB + kt + 56);
      if (kt + 64 < K) {  // deep prefetch (global_prefetch_b8 path)
        __builtin_prefetch(gpA + kt + 64, 0, 1);
        __builtin_prefetch(gpB + kt + 64, 0, 1);
      }
    }

    v16h a0 = load_frag(As, wm * 32, 32, lane);
    v16h a1 = load_frag(As, wm * 32 + 16, 32, lane);
#pragma unroll
    for (int ni = 0; ni < 4; ++ni) {
      v16h b = load_frag(Bts, wn * 64 + ni * 16, 32, lane);
      acc[0][ni] = WMMA_F16(a0, b, acc[0][ni]);
      acc[1][ni] = WMMA_F16(a1, b, acc[1][ni]);
    }
  }

  // Epilogue per CDNA5 C/D layout: elem r -> (M = r + 8*(lane/16), N = lane%16).
  const int colp = lane & 15, rowoff = (lane >> 4) * 8;
#pragma unroll
  for (int mi = 0; mi < 2; ++mi) {
#pragma unroll
    for (int ni = 0; ni < 4; ++ni) {
#pragma unroll
      for (int r = 0; r < 8; ++r) {
        int rr = m0 + wm * 32 + mi * 16 + rowoff + r;
        int cc = n0 + wn * 64 + ni * 16 + colp;
        float v = acc[mi][ni][r] + bias[cc];
        if (MODE == 1) {
          v = 0.5f * v * (1.0f + tanhf(0.7978845608f * (v + 0.044715f * v * v * v)));
        }
        if (MODE == 2) outf[(size_t)rr * N + cc] = v;
        else           outh[(size_t)rr * N + cc] = (half_t)v;
      }
    }
  }
}

// ---------------------------------------------------------------------------
// Attention per (window-batch, head): cosine attention + CPB bias + shift mask
// + softmax + P@V, all WMMA-driven. Block = 128 threads (4 waves).
// ---------------------------------------------------------------------------
__device__ __forceinline__ int mask_region(int win, int n) {
  int y = ((win >> 3) << 3) + (n >> 3);
  int x = ((win & 7) << 3) + (n & 7);
  int a = (y < 56) ? 0 : (y < 60) ? 1 : 2;
  int b = (x < 56) ? 0 : (x < 60) ? 1 : 2;
  return a * 3 + b;
}

__launch_bounds__(128)
__global__ void attn_kernel(const half_t* __restrict__ qkv, const float* __restrict__ tau,
                            const float* __restrict__ cpb, half_t* __restrict__ ao) {
  __shared__ alignas(16) half_t qs[64 * 32];
  __shared__ alignas(16) half_t ks[64 * 32];
  __shared__ alignas(16) half_t vt[32 * 64];   // transposed: vt[d][m]
  __shared__ float qn[64], kn[64];
  __shared__ float Sb[64 * 64];
  __shared__ alignas(16) half_t Pb[64 * 64];

  const int bw = blockIdx.x;          // 0..1023 (= b*64 + win)
  const int hh = blockIdx.y;          // head
  const int win = bw & 63;
  const int tid = threadIdx.x, lane = tid & 31, wv = tid >> 5;

  // Stage q, k row-major and v transposed.
  {
    int row = tid >> 1, kh = (tid & 1) * 16;
    const half_t* qp = qkv + (size_t)(bw * 64 + row) * 1536 + hh * HD + kh;
#pragma unroll
    for (int i = 0; i < 16; ++i) {
      qs[row * 32 + kh + i] = qp[i];
      ks[row * 32 + kh + i] = qp[512 + i];
      vt[(kh + i) * 64 + row] = qp[1024 + i];
    }
  }
  __syncthreads();

  if (tid < 64) {
    float sq = 0.0f, sk = 0.0f;
    for (int d = 0; d < HD; ++d) {
      float a = (float)qs[tid * 32 + d]; sq += a * a;
      float b = (float)ks[tid * 32 + d]; sk += b * b;
    }
    qn[tid] = sqrtf(sq);
    kn[tid] = sqrtf(sk);
  }
  __syncthreads();

  const float lscale = fmaxf(tau[hh] + 2.302585093f, 0.01f);
  const int colp = lane & 15, rowoff = (lane >> 4) * 8;

  // S = Q * K^T  (K = HD = 32 -> single WMMA k-step; 4 key tiles of 16).
  {
    v16h a = load_frag(qs, wv * 16, 32, lane);
#pragma unroll
    for (int nt = 0; nt < 4; ++nt) {
      v16h b = load_frag(ks, nt * 16, 32, lane);
      v8f c = {};
      c = WMMA_F16(a, b, c);
      int m = nt * 16 + colp;                         // key index
      int rm = mask_region(win, m);
#pragma unroll
      for (int r = 0; r < 8; ++r) {
        int n = wv * 16 + rowoff + r;                 // query index
        float denom = fmaxf(qn[n] * kn[m], 1e-6f);
        float v = c[r] * __builtin_amdgcn_rcpf(denom) * lscale;  // fast v_rcp_f32
        int idx = ((n >> 3) - (m >> 3) + 7) * 15 + ((n & 7) - (m & 7) + 7);
        v += cpb[idx * NHEAD + hh];
        if (mask_region(win, n) != rm) v += -100.0f;
        Sb[n * 64 + m] = v;
      }
    }
  }
  __syncthreads();

  // Softmax over keys; write P as f16.
  if (tid < 64) {
    float mx = -1e30f;
    for (int m = 0; m < 64; ++m) mx = fmaxf(mx, Sb[tid * 64 + m]);
    float sum = 0.0f;
    for (int m = 0; m < 64; ++m) {
      float e = __expf(Sb[tid * 64 + m] - mx);
      Sb[tid * 64 + m] = e;
      sum += e;
    }
    float inv = __builtin_amdgcn_rcpf(sum);
    for (int m = 0; m < 64; ++m) Pb[tid * 64 + m] = (half_t)(Sb[tid * 64 + m] * inv);
  }
  __syncthreads();

  // O = P @ V : M=64 queries, N=32 dims (2 tiles), K=64 keys (2 k-steps).
  v8f o0 = {}, o1 = {};
#pragma unroll
  for (int kk = 0; kk < 2; ++kk) {
    v16h ap = load_frag(Pb + kk * 32, wv * 16, 64, lane);
    v16h b0 = load_frag(vt + kk * 32, 0, 64, lane);
    v16h b1 = load_frag(vt + kk * 32, 16, 64, lane);
    o0 = WMMA_F16(ap, b0, o0);
    o1 = WMMA_F16(ap, b1, o1);
  }
#pragma unroll
  for (int r = 0; r < 8; ++r) {
    int n = wv * 16 + rowoff + r;
    size_t base = (size_t)(bw * 64 + n) * CDIM + hh * HD;
    ao[base + colp]      = (half_t)o0[r];
    ao[base + 16 + colp] = (half_t)o1[r];
  }
}

// ---------------------------------------------------------------------------
// Block-wide sum reduction (256 threads).
// ---------------------------------------------------------------------------
__device__ __forceinline__ float block_sum(float v, float* red) {
  int tid = threadIdx.x;
  red[tid] = v;
  __syncthreads();
  for (int s = 128; s > 0; s >>= 1) {
    if (tid < s) red[tid] += red[tid + s];
    __syncthreads();
  }
  float r = red[0];
  __syncthreads();
  return r;
}

// Window-reverse + reverse shift + x1 = shortcut + LN(proj_out).
__global__ void ln_add_kernel(const float* __restrict__ y, const float* __restrict__ x,
                              const float* __restrict__ g, const float* __restrict__ bt,
                              float* __restrict__ x1, half_t* __restrict__ x1h) {
  __shared__ float red[256];
  int t = blockIdx.x;                           // spatial token
  int b = t >> 12, hw = t & 4095;
  int hy = hw >> 6, wx = hw & 63;
  int hs = (hy - 4) & 63, ws2 = (wx - 4) & 63;  // shifted-frame coords
  int win = (hs >> 3) * 8 + (ws2 >> 3);
  int n = (hs & 7) * 8 + (ws2 & 7);
  const float* yp = y + (size_t)((b * 64 + win) * 64 + n) * CDIM;
  const float* xp = x + (size_t)t * CDIM;
  float s = 0.0f;
  for (int i = threadIdx.x; i < CDIM; i += 256) s += yp[i];
  float mean = block_sum(s, red) * (1.0f / CDIM);
  float v = 0.0f;
  for (int i = threadIdx.x; i < CDIM; i += 256) { float d = yp[i] - mean; v += d * d; }
  float var = block_sum(v, red) * (1.0f / CDIM);
  float inv = rsqrtf(var + 1e-6f);
  for (int i = threadIdx.x; i < CDIM; i += 256) {
    float val = xp[i] + (yp[i] - mean) * inv * g[i] + bt[i];
    x1[(size_t)t * CDIM + i] = val;
    x1h[(size_t)t * CDIM + i] = (half_t)val;
  }
}

// out = x1 + LN(h2), in place on d_out (which holds h2 from the mlp2 GEMM).
__global__ void final_kernel(float* __restrict__ io, const float* __restrict__ x1,
                             const float* __restrict__ g, const float* __restrict__ bt) {
  __shared__ float red[256];
  int t = blockIdx.x;
  float* hp = io + (size_t)t * CDIM;
  const float* xp = x1 + (size_t)t * CDIM;
  float s = 0.0f;
  for (int i = threadIdx.x; i < CDIM; i += 256) s += hp[i];
  float mean = block_sum(s, red) * (1.0f / CDIM);
  float v = 0.0f;
  for (int i = threadIdx.x; i < CDIM; i += 256) { float d = hp[i] - mean; v += d * d; }
  float var = block_sum(v, red) * (1.0f / CDIM);
  float inv = rsqrtf(var + 1e-6f);
  for (int i = threadIdx.x; i < CDIM; i += 256)
    hp[i] = xp[i] + (hp[i] - mean) * inv * g[i] + bt[i];
}

// ---------------------------------------------------------------------------
extern "C" void kernel_launch(void* const* d_in, const int* in_sizes, int n_in,
                              void* d_out, int out_size, void* d_ws, size_t ws_size,
                              hipStream_t stream) {
  const float* x      = (const float*)d_in[0];
  const float* q_w    = (const float*)d_in[1];
  const float* q_b    = (const float*)d_in[2];
  const float* k_w    = (const float*)d_in[3];
  const float* v_w    = (const float*)d_in[4];
  const float* v_b    = (const float*)d_in[5];
  const float* proj_w = (const float*)d_in[6];
  const float* proj_b = (const float*)d_in[7];
  const float* tau    = (const float*)d_in[8];
  const float* cpb_w1 = (const float*)d_in[9];
  const float* cpb_b1 = (const float*)d_in[10];
  const float* cpb_w2 = (const float*)d_in[11];
  const float* cpb_b2 = (const float*)d_in[12];
  const float* n1g    = (const float*)d_in[13];
  const float* n1b    = (const float*)d_in[14];
  const float* n2g    = (const float*)d_in[15];
  const float* n2b    = (const float*)d_in[16];
  const float* mlp_w1 = (const float*)d_in[17];
  const float* mlp_b1 = (const float*)d_in[18];
  const float* mlp_w2 = (const float*)d_in[19];
  const float* mlp_b2 = (const float*)d_in[20];

  char* w = (char*)d_ws;
  half_t* wqkvT  = (half_t*)w;  w += (size_t)512 * 1536 * 2;
  half_t* wprojT = (half_t*)w;  w += (size_t)512 * 512 * 2;
  half_t* wm1T   = (half_t*)w;  w += (size_t)512 * 2048 * 2;
  half_t* wm2T   = (half_t*)w;  w += (size_t)2048 * 512 * 2;
  float*  bqkv   = (float*)w;   w += 1536 * 4;
  float*  cpb    = (float*)w;   w += 16384;
  half_t* xw     = (half_t*)w;  w += (size_t)TOK * 512 * 2;
  half_t* qkvh   = (half_t*)w;  w += (size_t)TOK * 1536 * 2;
  half_t* aoh    = (half_t*)w;  w += (size_t)TOK * 512 * 2;
  float*  x1     = (float*)w;   w += (size_t)TOK * 512 * 4;
  half_t* x1h    = (half_t*)w;  w += (size_t)TOK * 512 * 2;
  // ybuf (proj out, f32, 128MB) aliases hh (mlp hidden, f16, 256MB): disjoint lifetimes.
  float*  ybuf   = (float*)w;
  half_t* hh     = (half_t*)w;

  pack_weights_kernel<<<2048, 256, 0, stream>>>(q_w, q_b, k_w, v_w, v_b, proj_w,
                                                mlp_w1, mlp_w2, wqkvT, bqkv, wprojT, wm1T, wm2T);
  shift_partition_kernel<<<TOK, 256, 0, stream>>>(x, xw);
  cpb_kernel<<<225, 256, 0, stream>>>(cpb_w1, cpb_b1, cpb_w2, cpb_b2, cpb);

  // Fused QKV GEMM: [65536 x 512] x [512 x 1536]
  gemm_kernel<0><<<dim3(1536 / 128, TOK / 64), 128, 0, stream>>>(
      xw, wqkvT, bqkv, nullptr, qkvh, TOK, 1536, 512);

  attn_kernel<<<dim3(1024, NHEAD), 128, 0, stream>>>(qkvh, tau, cpb, aoh);

  // Output projection -> f32
  gemm_kernel<2><<<dim3(512 / 128, TOK / 64), 128, 0, stream>>>(
      aoh, wprojT, proj_b, ybuf, nullptr, TOK, 512, 512);

  ln_add_kernel<<<TOK, 256, 0, stream>>>(ybuf, x, n1g, n1b, x1, x1h);

  // MLP fc1 + gelu -> f16
  gemm_kernel<1><<<dim3(2048 / 128, TOK / 64), 128, 0, stream>>>(
      x1h, wm1T, mlp_b1, nullptr, hh, TOK, 2048, 512);

  // MLP fc2 -> f32 directly into d_out (as h2)
  gemm_kernel<2><<<dim3(512 / 128, TOK / 64), 128, 0, stream>>>(
      hh, wm2T, mlp_b2, (float*)d_out, nullptr, TOK, 512, 2048);

  final_kernel<<<TOK, 256, 0, stream>>>((float*)d_out, x1, n2g, n2b);
}
